// SimpleAttention_13778255085650
// MI455X (gfx1250) — matrix operand order repaired and checked by reading, not compiled, not run on hardware
//
#include <hip/hip_runtime.h>
#include <hip/hip_bf16.h>

// ---------------------------------------------------------------------------
// Flash attention for B=8, S=4096, D=128 (fp32 in/out, bf16 WMMA compute).
//
// Sᵀ trick: score tiles are computed as K_tile(16x128) x Qᵀ(128x16) so the
// 16x16 f32 result (C layout: VGPR=row=key, lane=query) is directly the
// A-fragment layout (lane=M=query, K=key packed in regs) required for the
// P x V WMMA. No cross-lane transpose of P.
//
// Round-2 changes:
//  * double-buffered DS fragment loads (QK and PV phases) to hide LDS latency
//  * wave-uniform (readfirstlane) skip of the accumulator rescale when no
//    row max changed this tile -> removes 64 mul + 8 bpermute most iterations
//  * packed f32->bf16 conversions
// ---------------------------------------------------------------------------

typedef __attribute__((ext_vector_type(16))) __bf16 v16bf;
typedef __attribute__((ext_vector_type(8)))  float  v8f;

#define S_LEN   4096
#define D_DIM   128
#define BLK_M   128            // queries per block (8 waves x 16)
#define BLK_N   32             // keys per iteration
#define N_ITER  (S_LEN / BLK_N)

// scale * log2(e) : 1/sqrt(128) * 1.4426950408889634
#define SCALE_LOG2E 0.12751389635900512f

// LDS: K row-major bf16, rows padded 128->136 (272B, bank-spread, 16B-aligned)
//      V transposed bf16 [d][key], rows padded 32->40 (80B)
#define KPAD 136
#define VPAD 40

union V16 {
    v16bf    v;
    uint4    q[2];
    unsigned u[8];
};

__device__ __forceinline__ unsigned pack_bf2(float lo, float hi) {
    union { __bf16 h[2]; unsigned u; } c;
    c.h[0] = (__bf16)lo;
    c.h[1] = (__bf16)hi;
    return c.u;
}

__global__ __launch_bounds__(256) void
fa_fwd_kernel(const float* __restrict__ Q,
              const float* __restrict__ K,
              const float* __restrict__ V,
              float* __restrict__ O)
{
    __shared__ __align__(16) unsigned short Klds[BLK_N][KPAD];
    __shared__ __align__(16) unsigned short Vlds[D_DIM][VPAD];

    const int tid  = threadIdx.x;
    const int lane = tid & 31;
    const int wave = tid >> 5;
    const int ln   = lane & 15;     // N column / M row within tile
    const int g    = lane >> 4;     // lane half

    const int b      = blockIdx.x >> 5;        // 32 blocks per batch
    const int qblock = (blockIdx.x & 31) * BLK_M;

    const float* Qb = Q + (size_t)b * S_LEN * D_DIM;
    const float* Kb = K + (size_t)b * S_LEN * D_DIM;
    const float* Vb = V + (size_t)b * S_LEN * D_DIM;
    float*       Ob = O + (size_t)b * S_LEN * D_DIM;

    // ---- load Q once: B-fragments (lane = query column, dims 8g..8g+7 and 16+8g..16+8g+7 of each 32-chunk) ----
    const int qrow = qblock + wave * 16 + ln;   // this lane's query
    V16 qf[4];
#pragma unroll
    for (int j = 0; j < 4; ++j) {
        const float4* p =
            (const float4*)(Qb + (size_t)qrow * D_DIM + j * 32 + g * 8);
        float fs[16];
        *(float4*)&fs[0]  = p[0];
        *(float4*)&fs[4]  = p[1];
        *(float4*)&fs[8]  = p[4];
        *(float4*)&fs[12] = p[5];
#pragma unroll
        for (int i = 0; i < 8; ++i)
            qf[j].u[i] = pack_bf2(fs[2 * i], fs[2 * i + 1]);
    }

    // ---- accumulators / softmax state ----
    v8f acc[8];
#pragma unroll
    for (int t = 0; t < 8; ++t) acc[t] = (v8f)0.0f;
    float m_run = -__builtin_inff();
    float l_run = 0.0f;

    // ---- staging assignments for cooperative K/V tile loads ----
    const int kkey = tid >> 3;            // 0..31
    const int kd   = (tid & 7) * 16;      // 0..112
    const int vkp  = tid & 15;            // key pair 0..15
    const int vd   = (tid >> 4) * 8;      // 0..120

    float4 kr[4], vr[4];
    {   // prologue: tile 0
        const float4* pk = (const float4*)(Kb + (size_t)kkey * D_DIM + kd);
        kr[0] = pk[0]; kr[1] = pk[1]; kr[2] = pk[2]; kr[3] = pk[3];
        const float4* pv0 = (const float4*)(Vb + (size_t)(2 * vkp) * D_DIM + vd);
        const float4* pv1 = (const float4*)(Vb + (size_t)(2 * vkp + 1) * D_DIM + vd);
        vr[0] = pv0[0]; vr[1] = pv0[1];
        vr[2] = pv1[0]; vr[3] = pv1[1];
    }

    for (int it = 0; it < N_ITER; ++it) {
        __syncthreads();   // previous tile's LDS reads complete

        // ---- stage K tile (row-major bf16) ----
        {
            float fs[16];
            *(float4*)&fs[0]  = kr[0];
            *(float4*)&fs[4]  = kr[1];
            *(float4*)&fs[8]  = kr[2];
            *(float4*)&fs[12] = kr[3];
            unsigned u[8];
#pragma unroll
            for (int i = 0; i < 8; ++i) u[i] = pack_bf2(fs[2 * i], fs[2 * i + 1]);
            uint4* dst = (uint4*)&Klds[kkey][kd];
            dst[0] = *(uint4*)&u[0];
            dst[1] = *(uint4*)&u[4];
        }
        // ---- stage V tile transposed: Vlds[d][key] ----
        {
            float a0[8], a1[8];
            *(float4*)&a0[0] = vr[0]; *(float4*)&a0[4] = vr[1];
            *(float4*)&a1[0] = vr[2]; *(float4*)&a1[4] = vr[3];
#pragma unroll
            for (int i = 0; i < 8; ++i)
                *(unsigned*)&Vlds[vd + i][2 * vkp] = pack_bf2(a0[i], a1[i]);
        }
        __syncthreads();

        // ---- prefetch next tile into registers (overlaps with compute) ----
        if (it + 1 < N_ITER) {
            const int key0 = (it + 1) * BLK_N;
            const float4* pk =
                (const float4*)(Kb + (size_t)(key0 + kkey) * D_DIM + kd);
            kr[0] = pk[0]; kr[1] = pk[1]; kr[2] = pk[2]; kr[3] = pk[3];
            const float4* pv0 =
                (const float4*)(Vb + (size_t)(key0 + 2 * vkp) * D_DIM + vd);
            const float4* pv1 =
                (const float4*)(Vb + (size_t)(key0 + 2 * vkp + 1) * D_DIM + vd);
            vr[0] = pv0[0]; vr[1] = pv0[1];
            vr[2] = pv1[0]; vr[3] = pv1[1];
        }

        // ---- Sᵀ = K_tile x Qᵀ : two 16x16 tiles (key halves), 4 D-chunks ----
        // Double-buffered A-fragment loads: frag j+1 is in flight while the
        // WMMAs for frag j execute.
        v8f s0 = (v8f)0.0f, s1 = (v8f)0.0f;
        V16 afA[2], afB[2];
        afA[0].q[0] = *(const uint4*)&Klds[ln][8 * g];
        afA[0].q[1] = *(const uint4*)&Klds[ln][16 + 8 * g];
        afA[1].q[0] = *(const uint4*)&Klds[16 + ln][8 * g];
        afA[1].q[1] = *(const uint4*)&Klds[16 + ln][16 + 8 * g];
#pragma unroll
        for (int j = 0; j < 4; ++j) {
            V16* cur = (j & 1) ? afB : afA;
            V16* nxt = (j & 1) ? afA : afB;
            if (j < 3) {
                const int d0 = (j + 1) * 32 + 8 * g;
                nxt[0].q[0] = *(const uint4*)&Klds[ln][d0];
                nxt[0].q[1] = *(const uint4*)&Klds[ln][d0 + 16];
                nxt[1].q[0] = *(const uint4*)&Klds[16 + ln][d0];
                nxt[1].q[1] = *(const uint4*)&Klds[16 + ln][d0 + 16];
            }
            s0 = __builtin_amdgcn_wmma_f32_16x16x32_bf16(
                    false, cur[0].v, false, qf[j].v, (short)0, s0, false, false);
            s1 = __builtin_amdgcn_wmma_f32_16x16x32_bf16(
                    false, cur[1].v, false, qf[j].v, (short)0, s1, false, false);
        }

        // ---- online softmax (per lane: query = ln, 16 of 32 keys local) ----
        float mx = s0[0];
#pragma unroll
        for (int r = 0; r < 8; ++r) {
            mx = fmaxf(mx, s0[r]);
            mx = fmaxf(mx, s1[r]);
        }
        mx = fmaxf(mx, __shfl_xor(mx, 16, 32));

        // wave-uniform "did any row max change" flag -> scalar branch
        const int upd = __builtin_amdgcn_readfirstlane(
            (int)(__ballot(mx > m_run) != 0ull));

        const float mnew = fmaxf(m_run, mx);
        float p0[8], p1[8], sum = 0.0f;
#pragma unroll
        for (int r = 0; r < 8; ++r) {
            p0[r] = exp2f((s0[r] - mnew) * SCALE_LOG2E);
            p1[r] = exp2f((s1[r] - mnew) * SCALE_LOG2E);
            sum += p0[r] + p1[r];
        }
        sum += __shfl_xor(sum, 16, 32);

        // ---- P is already in A-fragment layout: just convert to bf16 ----
        V16 pf;
#pragma unroll
        for (int i = 0; i < 4; ++i) {
            pf.u[i]     = pack_bf2(p0[2 * i], p0[2 * i + 1]);
            pf.u[4 + i] = pack_bf2(p1[2 * i], p1[2 * i + 1]);
        }

        if (upd) {
            // rare path: row max moved -> rescale running state + accumulators
            const float alpha = exp2f((m_run - mnew) * SCALE_LOG2E);
            l_run = l_run * alpha + sum;
            m_run = mnew;
            float ar[8];
#pragma unroll
            for (int r = 0; r < 8; ++r) ar[r] = __shfl(alpha, r + 8 * g, 32);

            V16 vfA, vfB;
            vfA.q[0] = *(const uint4*)&Vlds[ln][g * 8];
            vfA.q[1] = *(const uint4*)&Vlds[ln][g * 8 + 16];
#pragma unroll
            for (int t = 0; t < 8; ++t) {
                V16& cur = (t & 1) ? vfB : vfA;
                V16& nxt = (t & 1) ? vfA : vfB;
                if (t < 7) {
                    const uint4* pv =
                        (const uint4*)&Vlds[(t + 1) * 16 + ln][g * 8];
                    nxt.q[0] = pv[0];
                    nxt.q[1] = pv[2];
                }
                v8f a = acc[t];
#pragma unroll
                for (int r = 0; r < 8; ++r) a[r] *= ar[r];
                acc[t] = __builtin_amdgcn_wmma_f32_16x16x32_bf16(
                            false, pf.v, false, cur.v, (short)0, a, false, false);
            }
        } else {
            // common path: alpha == 1 for every row
            l_run += sum;

            V16 vfA, vfB;
            vfA.q[0] = *(const uint4*)&Vlds[ln][g * 8];
            vfA.q[1] = *(const uint4*)&Vlds[ln][g * 8 + 16];
#pragma unroll
            for (int t = 0; t < 8; ++t) {
                V16& cur = (t & 1) ? vfB : vfA;
                V16& nxt = (t & 1) ? vfA : vfB;
                if (t < 7) {
                    const uint4* pv =
                        (const uint4*)&Vlds[(t + 1) * 16 + ln][g * 8];
                    nxt.q[0] = pv[0];
                    nxt.q[1] = pv[2];
                }
                acc[t] = __builtin_amdgcn_wmma_f32_16x16x32_bf16(
                            false, pf.v, false, cur.v, (short)0, acc[t],
                            false, false);
            }
        }
    }

    // ---- finalize: divide by row sums, write out ----
    float linv[8];
#pragma unroll
    for (int r = 0; r < 8; ++r)
        linv[r] = 1.0f / __shfl(l_run, r + 8 * g, 32);

#pragma unroll
    for (int t = 0; t < 8; ++t) {
#pragma unroll
        for (int r = 0; r < 8; ++r) {
            const int row = qblock + wave * 16 + r + 8 * g;
            Ob[(size_t)row * D_DIM + t * 16 + ln] = acc[t][r] * linv[r];
        }
    }
}

extern "C" void kernel_launch(void* const* d_in, const int* in_sizes, int n_in,
                              void* d_out, int out_size, void* d_ws, size_t ws_size,
                              hipStream_t stream)
{
    const float* Q = (const float*)d_in[0];
    const float* K = (const float*)d_in[1];
    const float* V = (const float*)d_in[2];
    float*       O = (float*)d_out;

    // 8 batches * (4096/128) query blocks = 256 blocks, 256 threads each
    dim3 grid(8 * (S_LEN / BLK_M));
    dim3 block(256);
    fa_fwd_kernel<<<grid, block, 0, stream>>>(Q, K, V, O);
}
